// Res_GCN_BCP_23845658427708
// MI455X (gfx1250) — compile-verified, matrix operand order checked by reading, not run on hardware
//
#include <hip/hip_runtime.h>
#include <math.h>

// ---------------------------------------------------------------------------
// Problem constants (from reference):  B=4, C=64, H=W=256, IC=32, SCALE=4
// hs=ws=64 -> N=4096, HW=65536.
// Algebraic collapse (matmul associativity + linearity of lerp):
//   M  = vr vr^T            [64x64]  per batch  (WMMA bf16)
//   S  = Phi M G^T + (Phi s) g_b^T + phi_b (G s)^T + N phi_b g_b^T   [32x32]
//   A  = Theta^T S / N      [64x32];  c = theta_b^T S / N  [32]
//   Pt = (A w_w^T)^T        [64x64] bf16 (stored [o][c]);  q = w_w c [64]
//   z  = vr^T P + q         [4096x64] per batch  (WMMA bf16)
//   out= v * (1 + sigmoid(0.1*(lerp_HW(z) + w_b)))
// Memory-bound: ~150 MB total traffic ≈ 6.5 us at 23.3 TB/s; the two GEMMs
// are cache-resident and ride the WMMA pipe.
// ---------------------------------------------------------------------------

#define NB  4
#define NC  64
#define NIC 32
#define NS  4096
#define HW  65536

typedef __attribute__((ext_vector_type(16))) __bf16 v16bf;
typedef __attribute__((ext_vector_type(8)))  float  v8f;

// K-index for element i of a 16-bit A/B fragment (doc 7.12.2), lane group g.
__device__ __forceinline__ int frag_k(int i, int g) {
  int vv = i >> 1, e = i & 1;
  return (vv < 4) ? (g * 8 + 2 * vv + e) : (16 + g * 8 + 2 * (vv - 4) + e);
}

// ---------------------------------------------------------------------------
// K1: bilinear 4x downsample (exact 2x2 average) -> vr (bf16) + row sums.
// One block per (b,c) row of vr.
// ---------------------------------------------------------------------------
__global__ void k_downsample(const float* __restrict__ v,
                             __bf16* __restrict__ vr,
                             float* __restrict__ srow) {
  int bc = blockIdx.x;                       // b*64 + c, 256 total
  const float* vp = v + (size_t)bc * (256 * 256);
  __bf16* op = vr + (size_t)bc * NS;
  int tid = threadIdx.x;
  float lsum = 0.f;
#pragma unroll
  for (int r = 0; r < 16; ++r) {
    int n = tid + r * 256;
    int h = n >> 6, w = n & 63;
    const float* p = vp + (4 * h + 1) * 256 + (4 * w + 1);
    float a = 0.25f * (p[0] + p[1] + p[256] + p[257]);
    op[n] = (__bf16)a;
    lsum += a;
  }
  __shared__ float red[256];
  red[tid] = lsum;
  __syncthreads();
  for (int off = 128; off > 0; off >>= 1) {
    if (tid < off) red[tid] += red[tid + off];
    __syncthreads();
  }
  if (tid == 0) srow[bc] = red[0];
}

// ---------------------------------------------------------------------------
// K2: Gram M = vr vr^T per batch via v_wmma_f32_16x16x32_bf16.
// 16 waves per block; wave w owns output tile (ti=w>>2, tj=w&3) and runs the
// full K=4096 (128 k-steps). One v8f accumulator per wave -> no spills, and
// each tile is exclusively owned -> direct global store, no reduction.
// ---------------------------------------------------------------------------
__global__ void k_gram(const __bf16* __restrict__ vr, float* __restrict__ M) {
  int b = blockIdx.x;
  const __bf16* vb = vr + (size_t)b * NC * NS;
  int tid = threadIdx.x;
  int lane = tid & 31, wave = tid >> 5;      // 16 waves
  int g = lane >> 4, m = lane & 15;
  int ti = wave >> 2, tj = wave & 3;

  const __bf16* ra = vb + (size_t)(ti * 16 + m) * NS;  // A rows
  const __bf16* rb = vb + (size_t)(tj * 16 + m) * NS;  // B "rows" (Gram: A^T)

  v8f acc = {};
#pragma unroll 2
  for (int ks = 0; ks < 128; ++ks) {
    int k0 = ks * 32;
    v16bf fa, fb;
#pragma unroll
    for (int i = 0; i < 16; ++i) {
      int kk = frag_k(i, g);
      fa[i] = ra[k0 + kk];
      fb[i] = rb[k0 + kk];
    }
    acc = __builtin_amdgcn_wmma_f32_16x16x32_bf16(
        false, fa, false, fb, (short)0, acc, false, false);
  }

  float* Mb = M + (size_t)b * NC * NC;
#pragma unroll
  for (int j = 0; j < 8; ++j) {
    int row = ti * 16 + j + 8 * g;           // C/D layout: VGPR j -> M=j (+8 hi)
    int col = tj * 16 + m;                   // lane%16 -> N
    Mb[row * NC + col] = acc[j];
  }
}

// ---------------------------------------------------------------------------
// K3: tiny closed-form middle math per batch (plain VALU; trivial FLOPs).
// Emits Pt in bf16, transposed [o][c] so K4's B-fragment gathers coalesce.
// ---------------------------------------------------------------------------
__global__ void k_small(const float* __restrict__ M, const float* __restrict__ srow,
                        const float* __restrict__ g_w, const float* __restrict__ g_b,
                        const float* __restrict__ theta_w, const float* __restrict__ theta_b,
                        const float* __restrict__ phi_w, const float* __restrict__ phi_b,
                        const float* __restrict__ w_w,
                        __bf16* __restrict__ Pt, float* __restrict__ q) {
  int b = blockIdx.x;
  const float* Mb = M + (size_t)b * NC * NC;
  const float* sb = srow + (size_t)b * NC;
  __shared__ float t1[NIC * NC];   // Phi * M      [32x64]
  __shared__ float S[NIC * NIC];   // [32x32]
  __shared__ float A[NC * NIC];    // [64x32]
  __shared__ float phis[NIC], gs[NIC], cvec[NIC];
  int tid = threadIdx.x;

  for (int o = tid; o < NIC * NC; o += 256) {
    int i = o >> 6, j = o & 63;
    float acc = 0.f;
    for (int c = 0; c < NC; ++c) acc += phi_w[i * NC + c] * Mb[c * NC + j];
    t1[o] = acc;
  }
  if (tid < NIC) {
    float a = 0.f, bb = 0.f;
    for (int c = 0; c < NC; ++c) {
      a  += phi_w[tid * NC + c] * sb[c];
      bb += g_w[tid * NC + c] * sb[c];
    }
    phis[tid] = a; gs[tid] = bb;
  }
  __syncthreads();

  for (int o = tid; o < NIC * NIC; o += 256) {
    int i = o >> 5, j = o & 31;
    float acc = 0.f;
    for (int c = 0; c < NC; ++c) acc += t1[i * NC + c] * g_w[j * NC + c];
    acc += phis[i] * g_b[j] + phi_b[i] * gs[j] + (float)NS * phi_b[i] * g_b[j];
    S[o] = acc;
  }
  __syncthreads();

  const float inv = 1.f / (float)NS;
  for (int o = tid; o < NC * NIC; o += 256) {
    int c = o >> 5, j = o & 31;
    float acc = 0.f;
    for (int i = 0; i < NIC; ++i) acc += theta_w[i * NC + c] * S[i * NIC + j];
    A[o] = acc * inv;
  }
  if (tid < NIC) {
    float acc = 0.f;
    for (int i = 0; i < NIC; ++i) acc += theta_b[i] * S[i * NIC + tid];
    cvec[tid] = acc * inv;
  }
  __syncthreads();

  __bf16* Ptb = Pt + (size_t)b * NC * NC;
  for (int o = tid; o < NC * NC; o += 256) {
    int c = o >> 6, oo = o & 63;             // c = K-dim (input ch), oo = out ch
    float acc = 0.f;
    for (int j = 0; j < NIC; ++j) acc += A[c * NIC + j] * w_w[oo * NIC + j];
    Ptb[oo * NC + c] = (__bf16)acc;          // transposed: [o][c]
  }
  if (tid < NC) {
    float acc = 0.f;
    for (int j = 0; j < NIC; ++j) acc += cvec[j] * w_w[tid * NIC + j];
    q[(size_t)b * NC + tid] = acc;
  }
}

// ---------------------------------------------------------------------------
// K4: z = vr^T P + q per batch via WMMA. Each wave -> one 16-row N-tile x
// 4 C-tiles (4 accumulators, 32 VGPRs), K=64 in two 32-wide steps.
// z stored [N][C] so K5 reads rows contiguously.
// ---------------------------------------------------------------------------
__global__ void k_apply(const __bf16* __restrict__ vr, const __bf16* __restrict__ Pt,
                        const float* __restrict__ q, float* __restrict__ z) {
  int b = blockIdx.x >> 5, blk = blockIdx.x & 31;
  const __bf16* vb = vr + (size_t)b * NC * NS;
  const __bf16* Ptb = Pt + (size_t)b * NC * NC;
  const float* qb = q + (size_t)b * NC;
  int tid = threadIdx.x, lane = tid & 31, wave = tid >> 5;
  int g = lane >> 4, m = lane & 15;
  int nbase = blk * 128 + wave * 16;         // 32 blocks * 8 waves * 16 = 4096

  v8f acc[4];
  v8f zero = {};
#pragma unroll
  for (int t = 0; t < 4; ++t) acc[t] = zero;

#pragma unroll
  for (int ks = 0; ks < 2; ++ks) {
    int k0 = ks * 32;
    v16bf fa;
    {                                         // A[n][k] = vr[k][n]
      const __bf16* cp = vb + nbase + m;
#pragma unroll
      for (int i = 0; i < 16; ++i)
        fa[i] = cp[(size_t)(k0 + frag_k(i, g)) * NS];
    }
#pragma unroll
    for (int t = 0; t < 4; ++t) {             // B[k][c] = Pt[c][k] (bf16)
      const __bf16* pp = Ptb + (size_t)(t * 16 + m) * NC + k0;
      v16bf fb;
#pragma unroll
      for (int i = 0; i < 16; ++i) fb[i] = pp[frag_k(i, g)];
      acc[t] = __builtin_amdgcn_wmma_f32_16x16x32_bf16(
          false, fa, false, fb, (short)0, acc[t], false, false);
    }
  }

  float* zb = z + (size_t)b * NS * NC;
#pragma unroll
  for (int t = 0; t < 4; ++t)
#pragma unroll
    for (int j = 0; j < 8; ++j) {
      int row = nbase + j + 8 * g;
      int col = t * 16 + m;
      zb[(size_t)row * NC + col] = acc[t][j] + qb[col];
    }
}

// ---------------------------------------------------------------------------
// K5: streaming finale. lerp z along flattened-N axis (scale 1/16), add w_b,
// sigmoid(0.1*x), residual gate. Fully coalesced on v/out; z rows hit L2.
// ---------------------------------------------------------------------------
__global__ void k_final(const float* __restrict__ v, const float* __restrict__ z,
                        const float* __restrict__ w_b, float* __restrict__ out) {
  size_t idx = (size_t)blockIdx.x * 256 + threadIdx.x;
  int j = (int)(idx & (HW - 1));
  int c = (int)((idx >> 16) & 63);
  int b = (int)(idx >> 22);
  const float* zb = z + (size_t)b * NS * NC;
  float cf = fminf(fmaxf((float)j * 0.0625f - 0.4375f, 0.f), (float)(NS - 1));
  int lo = (int)cf;
  int hi = min(lo + 1, NS - 1);
  float w = cf - (float)lo;
  float wy = (1.f - w) * zb[(size_t)lo * NC + c] + w * zb[(size_t)hi * NC + c] + w_b[c];
  float sg = 1.f / (1.f + __expf(-0.1f * wy));
  float vi = v[idx];
  out[idx] = vi * (1.f + sg);
}

// ---------------------------------------------------------------------------
extern "C" void kernel_launch(void* const* d_in, const int* in_sizes, int n_in,
                              void* d_out, int out_size, void* d_ws, size_t ws_size,
                              hipStream_t stream) {
  (void)in_sizes; (void)n_in; (void)out_size; (void)ws_size;
  const float* v       = (const float*)d_in[0];
  const float* g_w     = (const float*)d_in[1];
  const float* g_b     = (const float*)d_in[2];
  const float* theta_w = (const float*)d_in[3];
  const float* theta_b = (const float*)d_in[4];
  const float* phi_w   = (const float*)d_in[5];
  const float* phi_b   = (const float*)d_in[6];
  const float* w_w     = (const float*)d_in[7];
  const float* w_b     = (const float*)d_in[8];
  float* out = (float*)d_out;

  char* ws = (char*)d_ws;
  size_t ofs = 0;
  auto carve = [&](size_t bytes) -> void* {
    ofs = (ofs + 255) & ~(size_t)255;
    void* p = ws + ofs;
    ofs += bytes;
    return p;
  };
  __bf16* vr   = (__bf16*)carve((size_t)NB * NC * NS * sizeof(__bf16));  // 2 MB
  float*  srow = (float*) carve((size_t)NB * NC * sizeof(float));
  float*  M    = (float*) carve((size_t)NB * NC * NC * sizeof(float));
  __bf16* Pt   = (__bf16*)carve((size_t)NB * NC * NC * sizeof(__bf16));
  float*  q    = (float*) carve((size_t)NB * NC * sizeof(float));
  float*  z    = (float*) carve((size_t)NB * NS * NC * sizeof(float));   // 4 MB

  k_downsample<<<NB * NC, 256, 0, stream>>>(v, vr, srow);
  k_gram<<<NB, 512, 0, stream>>>(vr, M);
  k_small<<<NB, 256, 0, stream>>>(M, srow, g_w, g_b, theta_w, theta_b,
                                  phi_w, phi_b, w_w, Pt, q);
  k_apply<<<NB * 32, 256, 0, stream>>>(vr, Pt, q, z);
  k_final<<<(NB * NC * HW) / 256, 256, 0, stream>>>(v, z, w_b, out);
}
